// Conv1d_NN_Attn_44976897523806
// MI455X (gfx1250) — compile-verified, matrix operand order checked by reading, not run on hardware
//
#include <hip/hip_runtime.h>
#include <hip/hip_bf16.h>
#include <stdint.h>

#define TT 4096
#define CC 64
#define BB 4
#define TOPK 4
#define LDSTR 40   // bf16 elems per LDS row (32 used + pad); 80B => 16B-aligned, conflict-free

typedef __bf16 bf16t;
typedef bf16t v16bf __attribute__((ext_vector_type(16)));
typedef float  v8f   __attribute__((ext_vector_type(8)));

struct U256 { uint4 lo, hi; };
struct F16R { float4 v[4]; };   // 16 fp32 staged in registers

// fp32 -> bf16 round-to-nearest-even (bit trick)
__device__ __forceinline__ unsigned short f2bf(float f) {
    unsigned int u = __float_as_uint(f);
    return (unsigned short)((u + 0x7FFFu + ((u >> 16) & 1u)) >> 16);
}
__device__ __forceinline__ unsigned int pack2(float a, float b) {
    return (unsigned int)f2bf(a) | ((unsigned int)f2bf(b) << 16);
}

__device__ __forceinline__ F16R ld16(const float* p) {
    F16R r;
    r.v[0] = ((const float4*)p)[0];
    r.v[1] = ((const float4*)p)[1];
    r.v[2] = ((const float4*)p)[2];
    r.v[3] = ((const float4*)p)[3];
    return r;
}
__device__ __forceinline__ void st_lds16(unsigned short* d, const F16R& r) {
    ((uint4*)d)[0] = make_uint4(pack2(r.v[0].x, r.v[0].y), pack2(r.v[0].z, r.v[0].w),
                                pack2(r.v[1].x, r.v[1].y), pack2(r.v[1].z, r.v[1].w));
    ((uint4*)d)[1] = make_uint4(pack2(r.v[2].x, r.v[2].y), pack2(r.v[2].z, r.v[2].w),
                                pack2(r.v[3].x, r.v[3].y), pack2(r.v[3].z, r.v[3].w));
}

// Load one 16x32 bf16 WMMA fragment (A row-major / B column-major layout per
// CDNA5 ISA 7.12.2): lane holds row/col (lane&15); K elems {0..7,16..23}+8*(lane>>4),
// i.e. two contiguous 16B chunks at p and p+16 (p already includes +8*hb).
__device__ __forceinline__ v16bf frag_ld(const unsigned short* p) {
    U256 u;
    u.lo = *(const uint4*)(p);
    u.hi = *(const uint4*)(p + 16);
    return __builtin_bit_cast(v16bf, u);
}

// ---------------------------------------------------------------------------
// Generic projection GEMM: Out[m,n] = sum_k A[m,k] * W[n,k]
// A: (256 x 4096) fp32 row-major (B*C rows), W: (4096 x 4096) fp32 row-major.
// Macro-tile 128x128, 8 waves (each 32x64 = 2x4 WMMA tiles), K-step 32.
// Software-pipelined: global loads for step k+1 issued before the WMMAs of
// step k; double-buffered LDS; one barrier per k-step.
// ---------------------------------------------------------------------------
__global__ __launch_bounds__(256) void gemm_xwT(const float* __restrict__ A,
                                                const float* __restrict__ W,
                                                float* __restrict__ Out) {
    __shared__ unsigned short lA[2][128 * LDSTR];
    __shared__ unsigned short lB[2][128 * LDSTR];

    const int tid  = threadIdx.x;
    const int lane = tid & 31;
    const int w    = tid >> 5;
    const int wm   = w & 3;        // 4 wave-rows * 32 rows
    const int wn   = w >> 2;       // 2 wave-cols * 64 cols
    const int hb   = lane >> 4;
    const int l15  = lane & 15;
    const int m0   = blockIdx.y * 128;
    const int n0   = blockIdx.x * 128;
    const int lrow  = tid >> 1;          // 0..127 cooperative-load row
    const int lhalf = (tid & 1) * 16;    // 0 or 16 (k-half)
    const int ldsoff = lrow * LDSTR + lhalf;

    const float* pa = A + (size_t)(m0 + lrow) * TT + lhalf;
    const float* pb = W + (size_t)(n0 + lrow) * TT + lhalf;

    // prologue: stage k-step 0
    F16R ra = ld16(pa);
    F16R rb = ld16(pb);
    st_lds16(&lA[0][ldsoff], ra);
    st_lds16(&lB[0][ldsoff], rb);
    __syncthreads();

    v8f acc[2][4] = {};
    int cur = 0;

    for (int kk = 0; kk < TT; kk += 32) {
        const bool has_next = (kk + 32) < TT;
        // issue next tile's global loads first (latency hidden by WMMAs below)
        if (has_next) {
            ra = ld16(pa + kk + 32);
            rb = ld16(pb + kk + 32);
        }

        const unsigned short* bufA = lA[cur];
        const unsigned short* bufB = lB[cur];
        v16bf af[2], bfr[4];
        #pragma unroll
        for (int mi = 0; mi < 2; ++mi)
            af[mi] = frag_ld(bufA + (wm * 32 + mi * 16 + l15) * LDSTR + 8 * hb);
        #pragma unroll
        for (int ni = 0; ni < 4; ++ni)
            bfr[ni] = frag_ld(bufB + (wn * 64 + ni * 16 + l15) * LDSTR + 8 * hb);

        #pragma unroll
        for (int mi = 0; mi < 2; ++mi)
            #pragma unroll
            for (int ni = 0; ni < 4; ++ni)
                acc[mi][ni] = __builtin_amdgcn_wmma_f32_16x16x32_bf16(
                    false, af[mi], false, bfr[ni], (short)0, acc[mi][ni], false, false);

        // convert + park next tile in the alternate buffer
        if (has_next) {
            st_lds16(&lA[cur ^ 1][ldsoff], ra);
            st_lds16(&lB[cur ^ 1][ldsoff], rb);
        }
        __syncthreads();
        cur ^= 1;
    }

    // epilogue: D layout row = i + 8*hb, col = lane&15
    #pragma unroll
    for (int mi = 0; mi < 2; ++mi)
        #pragma unroll
        for (int ni = 0; ni < 4; ++ni) {
            const int r = m0 + wm * 32 + mi * 16 + 8 * hb;
            const int c = n0 + wn * 64 + ni * 16 + l15;
            float* o = Out + (size_t)r * TT + c;
            #pragma unroll
            for (int i = 0; i < 8; ++i)
                o[(size_t)i * TT] = acc[mi][ni][i];
        }
}

// ---------------------------------------------------------------------------
// Channel-L2-normalize (over C=64) + transpose (B,C,T) fp32 -> (B,T,C) bf16
// ---------------------------------------------------------------------------
__global__ __launch_bounds__(256) void norm_t(const float* __restrict__ src,
                                              unsigned short* __restrict__ dst) {
    const int b = blockIdx.y;
    const int t = blockIdx.x * 256 + threadIdx.x;
    const float* p = src + (size_t)b * CC * TT + t;
    float vals[CC];
    float ss = 0.f;
    #pragma unroll
    for (int c = 0; c < CC; ++c) {
        float v = p[(size_t)c * TT];
        vals[c] = v;
        ss += v * v;
    }
    float r = 1.0f / fmaxf(sqrtf(ss), 1e-12f);
    unsigned short* q = dst + ((size_t)b * TT + t) * CC;
    #pragma unroll
    for (int c = 0; c < CC; ++c) q[c] = f2bf(vals[c] * r);
}

// ---------------------------------------------------------------------------
// sim[t,s] = sum_c knT[t,c]*qnT[s,c]; relu; running top-4 per row.
// One wave per 16 t-rows. K=64 -> 2 bf16 WMMAs per 16x16 s-tile. Tile dumped
// to padded LDS; ALL 32 lanes scan: lane handles row (lane&15), columns
// hb*8..hb*8+7; the two half-lists are merged at the end (value desc,
// index asc on ties, matching lax.top_k).
// ---------------------------------------------------------------------------
__device__ __forceinline__ bool better(float v, int id, float tv, int ti) {
    return (v > tv) || (v == tv && id < ti);
}

__global__ __launch_bounds__(32) void sim_topk(const unsigned short* __restrict__ knT,
                                               const unsigned short* __restrict__ qnT,
                                               int* __restrict__ idx) {
    const int b    = blockIdx.y;
    const int tb   = blockIdx.x;                 // 16-row block of t
    const int lane = threadIdx.x;
    const int hb   = lane >> 4;
    const int l15  = lane & 15;

    const unsigned short* arow = knT + ((size_t)b * TT + tb * 16 + l15) * CC;
    v16bf af0 = frag_ld(arow + 8 * hb);          // c = 0..31
    v16bf af1 = frag_ld(arow + 32 + 8 * hb);     // c = 32..63

    __shared__ float tile[16 * 17];
    __shared__ float mv[16][TOPK];
    __shared__ int   mj[16][TOPK];

    float tv[TOPK] = {-1.f, -1.f, -1.f, -1.f};
    int   ti[TOPK] = {0, 0, 0, 0};

    const float* myrow = &tile[l15 * 17 + hb * 8];

    for (int st = 0; st < TT / 16; ++st) {
        const unsigned short* brow = qnT + ((size_t)b * TT + st * 16 + l15) * CC;
        v16bf bf0 = frag_ld(brow + 8 * hb);
        v16bf bf1 = frag_ld(brow + 32 + 8 * hb);
        v8f c = {};
        c = __builtin_amdgcn_wmma_f32_16x16x32_bf16(false, af0, false, bf0, (short)0, c, false, false);
        c = __builtin_amdgcn_wmma_f32_16x16x32_bf16(false, af1, false, bf1, (short)0, c, false, false);

        #pragma unroll
        for (int i = 0; i < 8; ++i)
            tile[(i + 8 * hb) * 17 + l15] = c[i];
        __syncthreads();

        #pragma unroll
        for (int j = 0; j < 8; ++j) {
            float v  = fmaxf(myrow[j], 0.f);          // relu before top_k
            int   id = st * 16 + hb * 8 + j;
            if (v > tv[3]) {                           // ids increase => strict >
                if (v > tv[2]) {
                    tv[3] = tv[2]; ti[3] = ti[2];
                    if (v > tv[1]) {
                        tv[2] = tv[1]; ti[2] = ti[1];
                        if (v > tv[0]) {
                            tv[1] = tv[0]; ti[1] = ti[0];
                            tv[0] = v; ti[0] = id;
                        } else { tv[1] = v; ti[1] = id; }
                    } else { tv[2] = v; ti[2] = id; }
                } else { tv[3] = v; ti[3] = id; }
            }
        }
        __syncthreads();
    }

    // merge the two per-row half-lists (lanes 16-31 publish, lanes 0-15 merge)
    if (lane >= 16) {
        #pragma unroll
        for (int j = 0; j < TOPK; ++j) { mv[l15][j] = tv[j]; mj[l15][j] = ti[j]; }
    }
    __syncthreads();
    if (lane < 16) {
        #pragma unroll
        for (int j = 0; j < TOPK; ++j) {
            float v  = mv[lane][j];
            int   id = mj[lane][j];
            if (better(v, id, tv[3], ti[3])) {
                if (better(v, id, tv[2], ti[2])) {
                    tv[3] = tv[2]; ti[3] = ti[2];
                    if (better(v, id, tv[1], ti[1])) {
                        tv[2] = tv[1]; ti[2] = ti[1];
                        if (better(v, id, tv[0], ti[0])) {
                            tv[1] = tv[0]; ti[1] = ti[0];
                            tv[0] = v; ti[0] = id;
                        } else { tv[1] = v; ti[1] = id; }
                    } else { tv[2] = v; ti[2] = id; }
                } else { tv[3] = v; ti[3] = id; }
            }
        }
        int* o = idx + ((size_t)b * TT + tb * 16 + lane) * TOPK;
        #pragma unroll
        for (int j = 0; j < TOPK; ++j) o[j] = ti[j];
    }
}

// ---------------------------------------------------------------------------
// y[b,c,t] = conv_b[c] + sum_{j,ci} conv_w[c,ci,j] * v[b,ci, idx[b,t,j]]
// Block handles 4 t values; gathered v columns staged in LDS.
// ---------------------------------------------------------------------------
__global__ __launch_bounds__(256) void gather_conv(const float* __restrict__ v,
                                                   const int* __restrict__ idx,
                                                   const float* __restrict__ cw,
                                                   const float* __restrict__ cb,
                                                   float* __restrict__ y) {
    __shared__ float g[4][TOPK][CC];   // [tloc][j][ci]
    const int b  = blockIdx.y;
    const int t0 = blockIdx.x * 4;
    const int tid = threadIdx.x;

    // cooperative gather: thread -> (tloc, j, 4 channels)
    {
        const int tloc = tid >> 6;
        const int j    = (tid >> 4) & 3;
        const int ci0  = (tid & 15) * 4;
        const int s    = idx[((size_t)b * TT + t0 + tloc) * TOPK + j];
        const float* vp = v + (size_t)b * CC * TT + s;
        #pragma unroll
        for (int q = 0; q < 4; ++q)
            g[tloc][j][ci0 + q] = vp[(size_t)(ci0 + q) * TT];
    }
    __syncthreads();

    const int c  = tid & 63;
    const int tl = tid >> 6;
    float acc = cb[c];
    const float* wrow = cw + c * CC * TOPK;        // conv_w[c, ci, j]
    #pragma unroll
    for (int j = 0; j < TOPK; ++j)
        #pragma unroll 8
        for (int ci = 0; ci < CC; ++ci)
            acc += wrow[ci * TOPK + j] * g[tl][j][ci];

    y[(size_t)b * CC * TT + (size_t)c * TT + t0 + tl] = acc;
}

// ---------------------------------------------------------------------------
extern "C" void kernel_launch(void* const* d_in, const int* in_sizes, int n_in,
                              void* d_out, int out_size, void* d_ws, size_t ws_size,
                              hipStream_t stream) {
    const float* x      = (const float*)d_in[0];
    const float* Wq     = (const float*)d_in[1];
    const float* Wk     = (const float*)d_in[2];
    const float* Wv     = (const float*)d_in[3];
    const float* Wo     = (const float*)d_in[4];
    const float* conv_w = (const float*)d_in[5];
    const float* conv_b = (const float*)d_in[6];

    char* ws = (char*)d_ws;
    float*          q   = (float*)(ws);                       //  4 MB
    float*          k   = (float*)(ws + ((size_t)4  << 20));  //  4 MB
    float*          v   = (float*)(ws + ((size_t)8  << 20));  //  4 MB
    float*          y   = (float*)(ws + ((size_t)12 << 20));  //  4 MB
    unsigned short* qnT = (unsigned short*)(ws + ((size_t)16 << 20)); // 2 MB
    unsigned short* knT = (unsigned short*)(ws + ((size_t)18 << 20)); // 2 MB
    int*            idx = (int*)(ws + ((size_t)20 << 20));            // 256 KB

    dim3 gemm_grid(TT / 128, (BB * CC) / 128);   // (32, 2)

    gemm_xwT<<<gemm_grid, 256, 0, stream>>>(x, Wq, q);
    gemm_xwT<<<gemm_grid, 256, 0, stream>>>(x, Wk, k);
    gemm_xwT<<<gemm_grid, 256, 0, stream>>>(x, Wv, v);

    norm_t<<<dim3(TT / 256, BB), 256, 0, stream>>>(q, qnT);
    norm_t<<<dim3(TT / 256, BB), 256, 0, stream>>>(k, knT);

    sim_topk<<<dim3(TT / 16, BB), 32, 0, stream>>>(knT, qnT, idx);

    gather_conv<<<dim3(TT / 4, BB), 256, 0, stream>>>(v, idx, conv_w, conv_b, y);

    gemm_xwT<<<gemm_grid, 256, 0, stream>>>(y, Wo, (float*)d_out);
}
